// SimpleRNN_22660247454194
// MI455X (gfx1250) — matrix operand order repaired and checked by reading, not compiled, not run on hardware
//
#include <hip/hip_runtime.h>
#include <hip/hip_bf16.h>

// ---------------------------------------------------------------------------
// GRU + RK4 scan for MI455X (gfx1250).
// Persistent 64-row batch tiles, all weights + recurrent state in LDS (bf16),
// GRU/head matmuls via v_wmma_f32_16x16x32_bf16, ODE integration in fp32 VALU.
// Round 2: fast transcendentals (v_rcp_f32 / v_exp_f32 based sigmoid & tanh)
// to shorten the serial per-step critical path; RK4 substeps fully unrolled.
// ---------------------------------------------------------------------------

typedef __attribute__((ext_vector_type(16))) __bf16 v16bf;
typedef __attribute__((ext_vector_type(8)))  __bf16 v8bf;
typedef __attribute__((ext_vector_type(8)))  float  v8f;

#define B_TOT   4096
#define KSTEPS  512
#define U_DIM   3
#define P_DIM   5
#define H_DIM   128
#define L_DIM   32
#define NSUB    10
#define RATE_LO 0.01f
#define RATE_HI 3.0f

#define BM       64      // batch rows per workgroup
#define NTHREADS 256     // 8 wave32s

// LDS strides (bf16 elements): multiples of 8 (=16B) for aligned ds b128 loads,
// chosen so 16 consecutive rows land in distinct bank groups.
#define WHH_S 136        // W_hh rows (384 x 128 used)
#define WIH_S 40         // W_ih rows (384 x 32 used)
#define HS    136        // h rows    (64  x 128 used)
#define XS    40         // x rows    (64  x 32 used)
#define HWS   136        // head_W rows (16 x 128, rows 8..15 zero)

#define N_WHH (384*WHH_S)
#define N_WIH (384*WIH_S)
#define N_WHD (16*HWS)
#define N_H   (2*BM*HS)
#define N_X   (BM*XS)
#define SMEM_BYTES ((size_t)((N_WHH+N_WIH+N_WHD+N_H+N_X)*2 + (L_DIM*8 + BM*8 + BM*8 + 16)*4))

// Fast transcendentals: v_rcp_f32 + v_exp_f32 (no IEEE division sequence).
__device__ __forceinline__ float fast_rcp(float x) { return __builtin_amdgcn_rcpf(x); }
__device__ __forceinline__ float sigm(float x) { return fast_rcp(1.0f + __expf(-x)); }
__device__ __forceinline__ float fast_tanh(float x) {
  // tanh(x) = 2*sigmoid(2x) - 1
  return __builtin_fmaf(2.0f, sigm(2.0f * x), -1.0f);
}

__device__ __forceinline__ v8f wmma_bf16(v16bf a, v16bf b, v8f c) {
  // D = A(16x32 bf16) * B(32x16 bf16) + C(16x16 f32)
  return __builtin_amdgcn_wmma_f32_16x16x32_bf16(false, a, false, b, (short)0, c, false, false);
}

// A-operand (16x32, MxK) from row-major LDS tile.
// ISA layout: VGPR0-3 lanes0-15 K=0..7 / lanes16-31 K=8..15; VGPR4-7 +16.
__device__ __forceinline__ v16bf load_tileA(const __bf16* __restrict__ base,
                                            int stride, int lane, int kbase) {
  const int row = lane & 15;
  const int ko  = (lane & 16) ? 8 : 0;
  const __bf16* p = base + row * stride + kbase + ko;
  v8bf lo = *(const v8bf*)(p);        // K = kbase+ko .. +7
  v8bf hi = *(const v8bf*)(p + 16);   // K = kbase+ko+16 .. +23
  return __builtin_shufflevector(lo, hi, 0,1,2,3,4,5,6,7,8,9,10,11,12,13,14,15);
}

// B-operand (32x16, KxN) where LDS holds W row-major (out-dim rows, contiguous K).
// ISA layout: lanes0-15 col=lane K=0..7,16..23; lanes16-31 col=lane-16 K=8..15,24..31.
__device__ __forceinline__ v16bf load_tileB(const __bf16* __restrict__ base,
                                            int stride, int lane, int nbase, int kbase) {
  const int col = lane & 15;
  const int ko  = (lane & 16) ? 8 : 0;
  const __bf16* p = base + (nbase + col) * stride + kbase + ko;
  v8bf lo = *(const v8bf*)(p);        // K = kbase+ko .. +7
  v8bf hi = *(const v8bf*)(p + 16);   // K = kbase+ko+16 .. +23
  return __builtin_shufflevector(lo, hi, 0,1,2,3,4,5,6,7,8,9,10,11,12,13,14,15);
}

__device__ __forceinline__ void rhs5(const float* __restrict__ y,
                                     const float* __restrict__ t,
                                     float* __restrict__ d) {
  const float A = y[0], D = y[1], G = y[2], J = y[3], M = y[4];
  const float f1 = t[0]*A, f2 = t[1]*D, f3 = t[2]*G, f4 = t[3]*J;
  const float r1 = t[4]*D, r2 = t[5]*G, r3 = t[6]*J, r4 = t[7]*M;
  d[0] = -f1 + r1;
  d[1] =  f1 - r1 - f2 + r2;
  d[2] =  f2 - r2 - f3 + r3;
  d[3] =  f3 - r3 - f4 + r4;
  d[4] =  f4 - r4;
}

__device__ __forceinline__ void rk4_step(float* __restrict__ y, float h,
                                         const float* __restrict__ t) {
  float k1[5], k2[5], k3[5], k4[5], tp[5];
  const float hh = 0.5f * h;
  rhs5(y, t, k1);
#pragma unroll
  for (int i = 0; i < 5; ++i) tp[i] = __builtin_fmaf(hh, k1[i], y[i]);
  rhs5(tp, t, k2);
#pragma unroll
  for (int i = 0; i < 5; ++i) tp[i] = __builtin_fmaf(hh, k2[i], y[i]);
  rhs5(tp, t, k3);
#pragma unroll
  for (int i = 0; i < 5; ++i) tp[i] = __builtin_fmaf(h, k3[i], y[i]);
  rhs5(tp, t, k4);
  const float h6 = h * (1.0f / 6.0f);
#pragma unroll
  for (int i = 0; i < 5; ++i)
    y[i] = fmaxf(__builtin_fmaf(h6, k1[i] + 2.0f*k2[i] + 2.0f*k3[i] + k4[i], y[i]), 0.0f);
}

__global__ __launch_bounds__(NTHREADS)
void gru_rk4_scan_kernel(const float* __restrict__ y0,
                         const float* __restrict__ u_seq,
                         const float* __restrict__ dt_seq,
                         const float* __restrict__ lift_W,
                         const float* __restrict__ lift_b,
                         const float* __restrict__ W_ih,
                         const float* __restrict__ W_hh,
                         const float* __restrict__ b_ih,
                         const float* __restrict__ b_hh,
                         const float* __restrict__ head_W,
                         const float* __restrict__ head_b,
                         const float* __restrict__ u_jump,
                         float* __restrict__ y_out,
                         float* __restrict__ th_out) {
  extern __shared__ __align__(16) char smem_raw[];
  __bf16* sWhh = (__bf16*)smem_raw;          // [384][WHH_S]
  __bf16* sWih = sWhh + N_WHH;               // [384][WIH_S]
  __bf16* sWhd = sWih + N_WIH;               // [16][HWS]
  __bf16* sH   = sWhd + N_WHD;               // [2][BM][HS]
  __bf16* sX   = sH   + N_H;                 // [BM][XS]
  float*  sLw  = (float*)(sX + N_X);         // [32][8]
  float*  sY   = sLw + L_DIM*8;              // [BM][8] (5 used)
  float*  sTh  = sY  + BM*8;                 // [BM][8]
  float*  sJmp = sTh + BM*8;                 // [16] (15 used)

  const int tid   = threadIdx.x;
  const int lane  = tid & 31;
  const int w     = tid >> 5;                // wave id 0..7
  const int bBase = blockIdx.x * BM;

  // ---- one-time init: weights -> bf16 LDS, state init ----
  for (int i = tid; i < 384*L_DIM; i += NTHREADS)
    sWih[(i >> 5) * WIH_S + (i & 31)] = (__bf16)W_ih[i];
  for (int i = tid; i < 384*H_DIM; i += NTHREADS)
    sWhh[(i >> 7) * WHH_S + (i & 127)] = (__bf16)W_hh[i];
  for (int i = tid; i < N_WHD; i += NTHREADS) sWhd[i] = (__bf16)0.0f;
  for (int i = tid; i < BM*HS; i += NTHREADS) sH[i] = (__bf16)0.0f;  // h0 = 0
  for (int i = tid; i < L_DIM*8; i += NTHREADS) sLw[i] = lift_W[i];
  if (tid < U_DIM*P_DIM) sJmp[tid] = u_jump[tid];
  for (int i = tid; i < BM*P_DIM; i += NTHREADS) {
    const int r = i / P_DIM, c = i - r*P_DIM;
    sY[r*8 + c] = y0[(size_t)(bBase + r)*P_DIM + c] + 0.01f;
  }
  __syncthreads();
  for (int i = tid; i < 8*H_DIM; i += NTHREADS)
    sWhd[(i >> 7) * HWS + (i & 127)] = (__bf16)head_W[i];

  // ---- loop-invariant per-thread registers ----
  float lb[8];
  {
    const int og = (tid & 3) * 8;
#pragma unroll
    for (int j = 0; j < 8; ++j) lb[j] = lift_b[og + j];
  }
  const int   hcol = (w << 4) | (lane & 15);      // h column this lane owns
  const float bR   = b_ih[hcol]           + b_hh[hcol];
  const float bZ   = b_ih[H_DIM + hcol]   + b_hh[H_DIM + hcol];
  const float bNi  = b_ih[2*H_DIM + hcol];
  const float bNh  = b_hh[2*H_DIM + hcol];
  const int   ccol = lane & 15;                   // C-layout column
  const int   rofs = (lane & 16) ? 8 : 0;         // C-layout row offset
  const float hbv  = (ccol < 8) ? head_b[ccol] : 0.0f;
  __syncthreads();

  int cur = 0;
#pragma unroll 1
  for (int k = 0; k < KSTEPS; ++k) {
    // ---------- Phase A: lift (VALU; 8 length-8 dots per thread) ----------
    {
      const int row = tid >> 2;
      const int og  = (tid & 3) * 8;
      const float* up = u_seq + ((size_t)(bBase + row)*KSTEPS + k)*U_DIM;
      const float f0 = up[0], f1 = up[1], f2 = up[2];
      const float yv0 = sY[row*8+0], yv1 = sY[row*8+1], yv2 = sY[row*8+2],
                  yv3 = sY[row*8+3], yv4 = sY[row*8+4];
#pragma unroll
      for (int j = 0; j < 8; ++j) {
        const float* wr = sLw + (og + j)*8;
        float a = wr[0]*f0 + wr[1]*f1 + wr[2]*f2 + wr[3]*yv0 + wr[4]*yv1
                + wr[5]*yv2 + wr[6]*yv3 + wr[7]*yv4 + lb[j];
        sX[row*XS + og + j] = (__bf16)(a * sigm(a));  // silu
      }
    }
    __syncthreads();

    // ---------- Phase B: GRU cell via WMMA (each wave owns 16 h-columns) ----------
    const __bf16* hcur = sH + cur*BM*HS;
    __bf16*       hnxt = sH + (cur^1)*BM*HS;
#pragma unroll 1
    for (int mt = 0; mt < 4; ++mt) {                // 4 M-tiles of 16 batch rows
      const __bf16* hrow = hcur + (mt*16)*HS;
      const v16bf ax  = load_tileA(sX + (mt*16)*XS, XS, lane, 0);
      const v16bf ah0 = load_tileA(hrow, HS, lane, 0);
      const v16bf ah1 = load_tileA(hrow, HS, lane, 32);
      const v16bf ah2 = load_tileA(hrow, HS, lane, 64);
      const v16bf ah3 = load_tileA(hrow, HS, lane, 96);
      const int nb = (w << 4);
      v8f accR  = {0.f,0.f,0.f,0.f,0.f,0.f,0.f,0.f};
      v8f accZ  = accR, accNi = accR, accNh = accR;
      // gi contributions (K=32)
      accR  = wmma_bf16(ax, load_tileB(sWih, WIH_S, lane, 0*H_DIM + nb, 0), accR);
      accZ  = wmma_bf16(ax, load_tileB(sWih, WIH_S, lane, 1*H_DIM + nb, 0), accZ);
      accNi = wmma_bf16(ax, load_tileB(sWih, WIH_S, lane, 2*H_DIM + nb, 0), accNi);
      // gh contributions (K=128 -> 4 chunks per gate)
      accR  = wmma_bf16(ah0, load_tileB(sWhh, WHH_S, lane, 0*H_DIM + nb,  0), accR);
      accR  = wmma_bf16(ah1, load_tileB(sWhh, WHH_S, lane, 0*H_DIM + nb, 32), accR);
      accR  = wmma_bf16(ah2, load_tileB(sWhh, WHH_S, lane, 0*H_DIM + nb, 64), accR);
      accR  = wmma_bf16(ah3, load_tileB(sWhh, WHH_S, lane, 0*H_DIM + nb, 96), accR);
      accZ  = wmma_bf16(ah0, load_tileB(sWhh, WHH_S, lane, 1*H_DIM + nb,  0), accZ);
      accZ  = wmma_bf16(ah1, load_tileB(sWhh, WHH_S, lane, 1*H_DIM + nb, 32), accZ);
      accZ  = wmma_bf16(ah2, load_tileB(sWhh, WHH_S, lane, 1*H_DIM + nb, 64), accZ);
      accZ  = wmma_bf16(ah3, load_tileB(sWhh, WHH_S, lane, 1*H_DIM + nb, 96), accZ);
      accNh = wmma_bf16(ah0, load_tileB(sWhh, WHH_S, lane, 2*H_DIM + nb,  0), accNh);
      accNh = wmma_bf16(ah1, load_tileB(sWhh, WHH_S, lane, 2*H_DIM + nb, 32), accNh);
      accNh = wmma_bf16(ah2, load_tileB(sWhh, WHH_S, lane, 2*H_DIM + nb, 64), accNh);
      accNh = wmma_bf16(ah3, load_tileB(sWhh, WHH_S, lane, 2*H_DIM + nb, 96), accNh);
      // gates + h update in C-layout (lane=col, vgpr=row)
#pragma unroll
      for (int j = 0; j < 8; ++j) {
        const float r = sigm(accR[j] + bR);
        const float z = sigm(accZ[j] + bZ);
        const float n = fast_tanh(accNi[j] + bNi + r*(accNh[j] + bNh));
        const int row = mt*16 + j + rofs;
        const float hold = (float)hcur[row*HS + hcol];
        hnxt[row*HS + hcol] = (__bf16)((1.0f - z)*n + z*hold);
      }
    }
    __syncthreads();

    // ---------- Phase C: head matmul + theta (waves 0..3, one M-tile each) ----------
    if (w < 4) {
      const __bf16* hrow = hnxt + (w*16)*HS;
      v8f acc = {0.f,0.f,0.f,0.f,0.f,0.f,0.f,0.f};
      acc = wmma_bf16(load_tileA(hrow, HS, lane,  0), load_tileB(sWhd, HWS, lane, 0,  0), acc);
      acc = wmma_bf16(load_tileA(hrow, HS, lane, 32), load_tileB(sWhd, HWS, lane, 0, 32), acc);
      acc = wmma_bf16(load_tileA(hrow, HS, lane, 64), load_tileB(sWhd, HWS, lane, 0, 64), acc);
      acc = wmma_bf16(load_tileA(hrow, HS, lane, 96), load_tileB(sWhd, HWS, lane, 0, 96), acc);
      if (ccol < 8) {
#pragma unroll
        for (int j = 0; j < 8; ++j) {
          const int row = w*16 + j + rofs;
          const float th = RATE_LO + (RATE_HI - RATE_LO) * sigm(acc[j] + hbv);
          sTh[row*8 + ccol] = th;
          th_out[((size_t)(bBase + row)*KSTEPS + k)*8 + ccol] = th;
        }
      }
    }
    __syncthreads();

    // ---------- Phase D: input jump + 10x RK4 (one batch row per lane) ----------
    if (tid < BM) {
      const int row = tid;
      const size_t bk = (size_t)(bBase + row)*KSTEPS + k;
      float th[8];
#pragma unroll
      for (int i = 0; i < 8; ++i) th[i] = sTh[row*8 + i];
      float y[5];
#pragma unroll
      for (int i = 0; i < 5; ++i) y[i] = sY[row*8 + i];
      const float* up = u_seq + bk*U_DIM;
      const float u0 = up[0], u1 = up[1], u2 = up[2];
#pragma unroll
      for (int c = 0; c < 5; ++c)
        y[c] += u0*sJmp[c] + u1*sJmp[5 + c] + u2*sJmp[10 + c];
      const float hsub = dt_seq[bk] * (1.0f / (float)NSUB);
#pragma unroll
      for (int s = 0; s < NSUB; ++s) rk4_step(y, hsub, th);
#pragma unroll
      for (int c = 0; c < 5; ++c) {
        sY[row*8 + c] = y[c];
        y_out[bk*P_DIM + c] = y[c];
      }
    }
    __syncthreads();
    cur ^= 1;
  }
}

extern "C" void kernel_launch(void* const* d_in, const int* in_sizes, int n_in,
                              void* d_out, int out_size, void* d_ws, size_t ws_size,
                              hipStream_t stream) {
  (void)in_sizes; (void)n_in; (void)out_size; (void)d_ws; (void)ws_size;
  const float* y0     = (const float*)d_in[0];
  const float* u_seq  = (const float*)d_in[1];
  const float* dt_seq = (const float*)d_in[2];
  const float* lift_W = (const float*)d_in[3];
  const float* lift_b = (const float*)d_in[4];
  const float* W_ih   = (const float*)d_in[5];
  const float* W_hh   = (const float*)d_in[6];
  const float* b_ih   = (const float*)d_in[7];
  const float* b_hh   = (const float*)d_in[8];
  const float* head_W = (const float*)d_in[9];
  const float* head_b = (const float*)d_in[10];
  const float* u_jump = (const float*)d_in[11];
  float* out    = (float*)d_out;
  float* y_out  = out;                                        // (B,K,5)
  float* th_out = out + (size_t)B_TOT*KSTEPS*P_DIM;           // (B,K,8)
  dim3 grid(B_TOT / BM), block(NTHREADS);
  hipLaunchKernelGGL(gru_rk4_scan_kernel, grid, block, SMEM_BYTES, stream,
                     y0, u_seq, dt_seq, lift_W, lift_b, W_ih, W_hh, b_ih, b_hh,
                     head_W, head_b, u_jump, y_out, th_out);
}
